// BloomAttention_14568529068079
// MI455X (gfx1250) — matrix operand order repaired and checked by reading, not compiled, not run on hardware
//
#include <hip/hip_runtime.h>
#include <cstddef>

// ---------------------------------------------------------------------------
// Types / helpers for gfx1250 WMMA (wave32)
// ---------------------------------------------------------------------------
typedef __attribute__((ext_vector_type(16))) __bf16 v16bf;
typedef __attribute__((ext_vector_type(8)))  __bf16 v8bf;
typedef __attribute__((ext_vector_type(8)))  float  v8f;

// Fragment viewed either as WMMA operand or as two 16-byte memory chunks.
union Frag16 {
    v16bf v;
    uint4 q[2];
};
union BF8Pack {
    v8bf  b;
    uint4 q;
    unsigned short u[8];
};

static __device__ __forceinline__ unsigned short f2bf(float f) {
    return (unsigned short)__builtin_bit_cast(unsigned short, (__bf16)f);
}

static __device__ __forceinline__ v8f zero8() {
    v8f z;
#pragma unroll
    for (int i = 0; i < 8; ++i) z[i] = 0.0f;
    return z;
}

static __device__ __forceinline__ v8f wmma_bf16(v16bf a, v16bf b, v8f c) {
    // (neg_a, A, neg_b, B, c_mod, C, reuse_a, reuse_b)
    return __builtin_amdgcn_wmma_f32_16x16x32_bf16(false, a, false, b, (short)0, c, false, false);
}

// A-matrix (16x32 bf16) fragment = two contiguous 8-elem chunks per lane:
//   elems 0..7  <- base[m*ld + h2*8 + 0..7]
//   elems 8..15 <- base[m*ld + 16 + h2*8 + 0..7]
static __device__ __forceinline__ v16bf load_a_frag(const unsigned short* base,
                                                    int ld, int lane) {
    const int m = lane & 15, h2 = lane >> 4;
    const unsigned short* p = base + (size_t)m * ld + h2 * 8;
    Frag16 f;
    f.q[0] = *(const uint4*)(p);
    f.q[1] = *(const uint4*)(p + 16);
    return f.v;
}
// B-matrix (32x16 bf16) fragment, from K-contiguous storage (BT[n][kk]):
//   elems 0..7 <- base[n*ld + h2*8 + 0..7], elems 8..15 <- base[n*ld + 16 + h2*8 + 0..7]
static __device__ __forceinline__ v16bf load_b_frag(const unsigned short* base,
                                                    int ld, int lane) {
    const int n = lane & 15, h2 = lane >> 4;
    const unsigned short* p = base + (size_t)n * ld + h2 * 8;
    Frag16 f;
    f.q[0] = *(const uint4*)(p);
    f.q[1] = *(const uint4*)(p + 16);
    return f.v;
}

// ---------------------------------------------------------------------------
// GEMM + bias:  C[M,N] = A[M,K] * Bw[K,N] + bias[N]   (N, K, SDIM compile-time)
//  MODE 0: C fp32
//  MODE 2: QKV mode — N = 3H; cols [0,2H) -> bf16 qk_ws [M,2H];
//          cols [2H,3H) -> bf16 vT [B*H, SDIM] transposed (row b*H+d, col s)
// Block: 256 threads = 8 waves (2x4); block tile 64x128, wave tile 32x32,
// KT=32, double-buffered LDS, register-staged software pipeline.
// ---------------------------------------------------------------------------
template <typename AT, int MODE, int N, int K, int SDIM>
__global__ __launch_bounds__(256) void gemm_bias_kernel(
    const AT* __restrict__ A, const float* __restrict__ Bw,
    const float* __restrict__ bias, void* __restrict__ Cout,
    unsigned short* __restrict__ Vout, int M)
{
    __shared__ alignas(16) unsigned short As[2][64][40];    // 64 x 32 (+pad)
    __shared__ alignas(16) unsigned short BsT[2][128][40];  // 128 cols x 32 k (T)

    const int t    = threadIdx.x;
    const int wave = t >> 5;
    const int lane = t & 31;
    const int waveM = wave >> 2;   // 0..1
    const int waveN = wave & 3;    // 0..3
    const int mBase = blockIdx.y * 64;
    const int nBase = blockIdx.x * 128;

    const int arow = t >> 2;           // 0..63 (4 threads per row)
    const int acol = (t & 3) * 8;      // 8 elems each (16B/32B)
    const int brow = t >> 3;           // 0..31 (8 threads per row)
    const int bcol = (t & 7) * 16;     // 16 elems each

    struct Stage { uint4 a; BF8Pack b0, b1; };

    auto load_stage = [&](int k0) {
        Stage s;
        if constexpr (sizeof(AT) == 4) {
            v8f af = *(const v8f*)((const float*)A +
                                   (size_t)(mBase + arow) * K + k0 + acol);
            BF8Pack p; p.b = __builtin_convertvector(af, v8bf);
            s.a = p.q;
        } else {
            s.a = *(const uint4*)((const unsigned short*)A +
                                  (size_t)(mBase + arow) * K + k0 + acol);
        }
        const float* bp = Bw + (size_t)(k0 + brow) * N + nBase + bcol;
        s.b0.b = __builtin_convertvector(*(const v8f*)(bp), v8bf);
        s.b1.b = __builtin_convertvector(*(const v8f*)(bp + 8), v8bf);
        return s;
    };
    auto store_stage = [&](int buf, const Stage& s) {
        *(uint4*)&As[buf][arow][acol] = s.a;
#pragma unroll
        for (int i = 0; i < 8; ++i) BsT[buf][bcol + i][brow]     = s.b0.u[i];
#pragma unroll
        for (int i = 0; i < 8; ++i) BsT[buf][bcol + 8 + i][brow] = s.b1.u[i];
    };

    v8f acc00 = zero8(), acc01 = zero8(), acc10 = zero8(), acc11 = zero8();

    Stage st = load_stage(0);
    store_stage(0, st);
    int cur = 0;

    for (int k0 = 0; k0 < K; k0 += 32) {
        __syncthreads();
        const bool more = (k0 + 32) < K;
        Stage nx;
        if (more) nx = load_stage(k0 + 32);   // issue next-chunk global loads early

        v16bf a0 = load_a_frag(&As[cur][waveM * 32 +  0][0], 40, lane);
        v16bf a1 = load_a_frag(&As[cur][waveM * 32 + 16][0], 40, lane);
        v16bf b0 = load_b_frag(&BsT[cur][waveN * 32 +  0][0], 40, lane);
        v16bf b1 = load_b_frag(&BsT[cur][waveN * 32 + 16][0], 40, lane);

        acc00 = wmma_bf16(a0, b0, acc00);
        acc01 = wmma_bf16(a0, b1, acc01);
        acc10 = wmma_bf16(a1, b0, acc10);
        acc11 = wmma_bf16(a1, b1, acc11);

        if (more) store_stage(cur ^ 1, nx);   // commit after compute
        cur ^= 1;
    }

    // Epilogue: C/D layout: lane (n = lane&15, h2 = lane>>4), VGPR r -> row r + 8*h2
    const int n  = lane & 15;
    const int h2 = lane >> 4;
#pragma unroll
    for (int ti = 0; ti < 2; ++ti) {
#pragma unroll
        for (int tj = 0; tj < 2; ++tj) {
            const v8f& acc = (ti == 0) ? (tj == 0 ? acc00 : acc01)
                                       : (tj == 0 ? acc10 : acc11);
            const int colG = nBase + waveN * 32 + tj * 16 + n;
            const float bv = bias[colG];
#pragma unroll
            for (int r = 0; r < 8; ++r) {
                const int rowG = mBase + waveM * 32 + ti * 16 + r + 8 * h2;
                const float val = acc[r] + bv;
                if constexpr (MODE == 0) {
                    ((float*)Cout)[(size_t)rowG * N + colG] = val;
                } else {
                    constexpr int H    = N / 3;
                    constexpr int twoH = 2 * H;
                    if (colG < twoH) {
                        ((unsigned short*)Cout)[(size_t)rowG * twoH + colG] = f2bf(val);
                    } else {
                        const int d = colG - twoH;                     // 0..H-1
                        const int b = rowG / SDIM, s = rowG - b * SDIM;
                        Vout[(size_t)(b * H + d) * SDIM + s] = f2bf(val);
                    }
                }
            }
        }
    }
}

// ---------------------------------------------------------------------------
// Fused ALiBi causal flash attention (bf16 in, bf16 out, f32 softmax).
// qk:  [B*S, 2H] bf16 (Q|K).  vT: [B*H, S] bf16.  ctx: [B*S, H] bf16.
// Block = 128 threads = 4 waves; each wave owns one 16-row query tile.
// grid = (B*nh, S/64).  S, H, NH compile-time.
// ---------------------------------------------------------------------------
template <int S, int H, int NH>
__global__ __launch_bounds__(128) void alibi_attn_kernel(
    const unsigned short* __restrict__ qk, const unsigned short* __restrict__ vT,
    unsigned short* __restrict__ ctx)
{
    constexpr int hd   = H / NH;   // 128
    constexpr int twoH = 2 * H;

    __shared__ alignas(16) unsigned short Ps[4][16 * 32];  // per-wave P (16 q x 32 keys)

    const int wave = threadIdx.x >> 5;
    const int lane = threadIdx.x & 31;
    const int ln   = lane & 15;
    const int h2   = lane >> 4;

    const int bh = blockIdx.x;
    const int b  = bh / NH;
    const int h  = bh % NH;
    const int qt = blockIdx.y * 4 + wave;   // query tile (16 rows)
    const int q0 = qt * 16;

    const unsigned short* qbase = qk + (size_t)(b * S) * twoH + h * hd;
    const unsigned short* kbase = qbase + H;
    const unsigned short* vTb   = vT + (size_t)(b * H + h * hd) * S;

    const float slope   = exp2f(-0.25f * (float)(h + 1));  // NH=32 power of two
    const float scaling = 0.08838834764831845f;            // 128^-0.5

    // Q fragments for the 4 hd-chunks (loaded once), direct 128-bit loads
    v16bf qf[4];
    {
        const unsigned short* qrow = qbase + (size_t)(q0 + ln) * twoH;
#pragma unroll
        for (int c = 0; c < 4; ++c) {
            Frag16 f;
            const unsigned short* p = qrow + c * 32 + h2 * 8;
            f.q[0] = *(const uint4*)(p);
            f.q[1] = *(const uint4*)(p + 16);
            qf[c] = f.v;
        }
    }

    v8f o[8];
#pragma unroll
    for (int t = 0; t < 8; ++t) o[t] = zero8();
    float mrow[8], lrow[8];
#pragma unroll
    for (int r = 0; r < 8; ++r) { mrow[r] = -1e30f; lrow[r] = 0.0f; }

    unsigned short* myP = Ps[wave];

    for (int jt = 0; jt <= qt; jt += 2) {
        const bool has2 = (jt + 1) <= qt;
        const int key0 = jt * 16;

        // ---- S = Q * K^T (two 16-key tiles); K rows are hd-contiguous ----
        v8f s0 = zero8(), s1 = zero8();
        {
            const unsigned short* krow = kbase + (size_t)(key0 + ln) * twoH + h2 * 8;
#pragma unroll
            for (int c = 0; c < 4; ++c) {
                Frag16 f;
                f.q[0] = *(const uint4*)(krow + c * 32);
                f.q[1] = *(const uint4*)(krow + c * 32 + 16);
                s0 = wmma_bf16(qf[c], f.v, s0);
            }
        }
        if (has2) {
            const unsigned short* krow = kbase + (size_t)(key0 + 16 + ln) * twoH + h2 * 8;
#pragma unroll
            for (int c = 0; c < 4; ++c) {
                Frag16 f;
                f.q[0] = *(const uint4*)(krow + c * 32);
                f.q[1] = *(const uint4*)(krow + c * 32 + 16);
                s1 = wmma_bf16(qf[c], f.v, s1);
            }
        }

        // ---- ALiBi + causal mask + online softmax ----
        float pr0[8], pr1[8];
#pragma unroll
        for (int r = 0; r < 8; ++r) {
            const int qrow = q0 + r + 8 * h2;
            const int kc0  = key0 + ln;
            float a0 = s0[r] * scaling + slope * (float)(kc0 - qrow);
            if (kc0 > qrow) a0 = -1e30f;
            float a1 = -1e30f;
            if (has2) {
                const int kc1 = key0 + 16 + ln;
                a1 = s1[r] * scaling + slope * (float)(kc1 - qrow);
                if (kc1 > qrow) a1 = -1e30f;
            }
            float cm = fmaxf(a0, a1);
#pragma unroll
            for (int off = 8; off >= 1; off >>= 1)
                cm = fmaxf(cm, __shfl_xor(cm, off, 32));
            const float nm = fmaxf(mrow[r], cm);
            const float p0 = __expf(a0 - nm);
            const float p1 = __expf(a1 - nm);
            float rs = p0 + p1;
#pragma unroll
            for (int off = 8; off >= 1; off >>= 1)
                rs += __shfl_xor(rs, off, 32);
            const float alpha = __expf(mrow[r] - nm);
            mrow[r] = nm;
            lrow[r] = lrow[r] * alpha + rs;
#pragma unroll
            for (int t = 0; t < 8; ++t) o[t][r] *= alpha;
            pr0[r] = p0; pr1[r] = p1;
        }

        // ---- relayout P (C-layout) -> A-fragment via per-wave LDS ----
#pragma unroll
        for (int r = 0; r < 8; ++r) {
            const int row = r + 8 * h2;
            myP[row * 32 + ln]      = f2bf(pr0[r]);
            myP[row * 32 + 16 + ln] = f2bf(pr1[r]);
        }
        asm volatile("s_wait_dscnt 0" ::: "memory");
        __builtin_amdgcn_wave_barrier();

        v16bf pf;
        {
            Frag16 f;
            const unsigned short* p = myP + ln * 32 + h2 * 8;
            f.q[0] = *(const uint4*)(p);
            f.q[1] = *(const uint4*)(p + 16);
            pf = f.v;
        }

        // ---- O += P * V ; vT rows are key-contiguous ----
#pragma unroll
        for (int t = 0; t < 8; ++t) {
            Frag16 f;
            const unsigned short* p = vTb + (size_t)(t * 16 + ln) * S + key0 + h2 * 8;
            f.q[0] = *(const uint4*)(p);
            f.q[1] = *(const uint4*)(p + 16);
            o[t] = wmma_bf16(pf, f.v, o[t]);
        }
    }

    // ---- normalize and store context (bf16) ----
    float inv[8];
#pragma unroll
    for (int r = 0; r < 8; ++r) inv[r] = 1.0f / lrow[r];
#pragma unroll
    for (int t = 0; t < 8; ++t)
#pragma unroll
        for (int r = 0; r < 8; ++r) {
            const int row = q0 + r + 8 * h2;
            const int col = h * hd + t * 16 + ln;
            ctx[(size_t)(b * S + row) * H + col] = f2bf(o[t][r] * inv[r]);
        }
}

// ---------------------------------------------------------------------------
// Launch
// ---------------------------------------------------------------------------
extern "C" void kernel_launch(void* const* d_in, const int* in_sizes, int n_in,
                              void* d_out, int out_size, void* d_ws, size_t ws_size,
                              hipStream_t stream) {
    const float* hidden = (const float*)d_in[0];  // [B,S,H]
    const float* w_qkv  = (const float*)d_in[1];  // [H,3H]
    const float* b_qkv  = (const float*)d_in[2];  // [3H]
    const float* w_out  = (const float*)d_in[3];  // [H,H]
    const float* b_out  = (const float*)d_in[4];  // [H]
    float* out = (float*)d_out;

    constexpr int H  = 4096, S = 1024, NH = 32;
    const int M = in_sizes[0] / H;   // B*S = 2048
    const int B = M / S;             // 2

    unsigned short* qk_ws  = (unsigned short*)d_ws;                 // [M, 2H] bf16
    unsigned short* vT_ws  = qk_ws + (size_t)M * 2 * H;             // [B*H, S] bf16
    unsigned short* ctx_ws = vT_ws + (size_t)B * H * S;             // [M, H]  bf16

    // 1) QKV projection: fp32 A -> bf16 qk workspace + transposed V
    gemm_bias_kernel<float, 2, 3 * H, H, S>
        <<<dim3((3 * H) / 128, M / 64), 256, 0, stream>>>(
            hidden, w_qkv, b_qkv, (void*)qk_ws, vT_ws, M);

    // 2) ALiBi causal flash attention
    alibi_attn_kernel<S, H, NH><<<dim3(B * NH, S / 64), 128, 0, stream>>>(
        qk_ws, vT_ws, ctx_ws);

    // 3) Output projection: bf16 A -> fp32 out with bias
    gemm_bias_kernel<unsigned short, 0, H, H, S>
        <<<dim3(H / 128, M / 64), 256, 0, stream>>>(
            ctx_ws, w_out, b_out, (void*)out, (unsigned short*)nullptr, M);
}
